// Model_18854906429631
// MI455X (gfx1250) — matrix operand order repaired and checked by reading, not compiled, not run on hardware
//
#include <hip/hip_runtime.h>

#define B_    8
#define N_    896
#define M_    512
#define ENF2  3072
#define AAD   1280
#define H_    32
#define D_    64
#define INNER 2048
#define CTX_  256

typedef __attribute__((ext_vector_type(16))) __bf16 v16bf;
typedef __attribute__((ext_vector_type(8)))  __bf16 v8bf;
typedef __attribute__((ext_vector_type(8)))  float  v8f;
typedef __attribute__((ext_vector_type(4)))  unsigned int v4u;
typedef __attribute__((ext_vector_type(4)))  int v4i;
typedef __attribute__((ext_vector_type(8)))  int v8i;

__device__ __forceinline__ unsigned short f2bf(float f) {
  unsigned int u = __float_as_uint(f);
  u += 0x7FFFu + ((u >> 16) & 1u);           // round-to-nearest-even
  return (unsigned short)(u >> 16);
}

__device__ __forceinline__ v16bf cat16(v8bf lo, v8bf hi) {
  return __builtin_shufflevector(lo, hi, 0,1,2,3,4,5,6,7,8,9,10,11,12,13,14,15);
}

// ---------------------------------------------------------------- conversions
__global__ void k_f32_to_bf16(const float* __restrict__ in,
                              unsigned short* __restrict__ out, size_t n) {
  for (size_t i = (size_t)blockIdx.x * blockDim.x + threadIdx.x; i < n;
       i += (size_t)gridDim.x * blockDim.x)
    out[i] = f2bf(in[i]);
}

// in: R x C row-major  ->  out: C x R row-major (i.e. W^T, K-contiguous per col)
__global__ void k_f32_to_bf16_T(const float* __restrict__ in,
                                unsigned short* __restrict__ out, int R, int C) {
  size_t n = (size_t)R * C;
  for (size_t o = (size_t)blockIdx.x * blockDim.x + threadIdx.x; o < n;
       o += (size_t)gridDim.x * blockDim.x) {
    int c = (int)(o / (size_t)R);
    int r = (int)(o % (size_t)R);
    out[o] = f2bf(in[(size_t)r * C + c]);
  }
}

// ------------------------------------------- GEMM + bias + per-head l2norm
// C(rows x INNER) = A(rows x K) * W^T(INNER x K)^T + bias, then l2norm over
// each 64-wide head segment, bf16 store to (b, h, row_in_batch, d).
// Block: 256 thr = 8 waves; block tile 16 rows x 256 cols (4 heads);
// each wave owns a 16x32 tile (2 accumulators sharing one A fragment).
__global__ void k_gemm_bias_l2norm(const __bf16* __restrict__ A,
                                   const __bf16* __restrict__ Wt,
                                   const float* __restrict__ bias,
                                   unsigned short* __restrict__ out_lat,
                                   int rows_per_batch, int K) {
  __shared__ float ssq[16][8];
  const int wave = threadIdx.x >> 5;
  const int lane = threadIdx.x & 31;
  const int half = lane >> 4;
  const int lc   = lane & 15;
  const int row0 = blockIdx.x * 16;
  const int col0 = blockIdx.y * 256 + wave * 32;

  const __bf16* ap  = A  + (size_t)(row0 + lc) * K + half * 8;
  const __bf16* bp0 = Wt + (size_t)(col0 + lc) * K + half * 8;
  const __bf16* bp1 = bp0 + (size_t)16 * K;

  v8f acc0 = {}, acc1 = {};
  for (int k0 = 0; k0 < K; k0 += 32) {
    v16bf a  = cat16(*(const v8bf*)(ap  + k0), *(const v8bf*)(ap  + k0 + 16));
    v16bf b0 = cat16(*(const v8bf*)(bp0 + k0), *(const v8bf*)(bp0 + k0 + 16));
    v16bf b1 = cat16(*(const v8bf*)(bp1 + k0), *(const v8bf*)(bp1 + k0 + 16));
    __builtin_prefetch(ap  + k0 + 32, 0, 3);
    __builtin_prefetch(bp0 + k0 + 32, 0, 3);
    __builtin_prefetch(bp1 + k0 + 32, 0, 3);
    acc0 = __builtin_amdgcn_wmma_f32_16x16x32_bf16(false, a, false, b0,
                                                   (short)0, acc0, false, false);
    acc1 = __builtin_amdgcn_wmma_f32_16x16x32_bf16(false, a, false, b1,
                                                   (short)0, acc1, false, false);
  }

  const float bv0 = bias[col0 + lc];
  const float bv1 = bias[col0 + 16 + lc];
  float v0[8], v1[8], ss[8];
#pragma unroll
  for (int v = 0; v < 8; ++v) {
    v0[v] = acc0[v] + bv0;
    v1[v] = acc1[v] + bv1;
    ss[v] = v0[v] * v0[v] + v1[v] * v1[v];
  }

  // row-wise sum of squares across the 16 lanes of this half
#pragma unroll
  for (int m = 1; m <= 8; m <<= 1)
#pragma unroll
    for (int v = 0; v < 8; ++v) ss[v] += __shfl_xor(ss[v], m, 32);

  if (lc == 0) {
#pragma unroll
    for (int v = 0; v < 8; ++v) ssq[v + 8 * half][wave] = ss[v];
  }
  __syncthreads();

  const int hb = wave & ~1;         // head = 64 cols = 2 waves
  const int h  = col0 >> 6;
  const int d0 = (col0 & 63) + lc;
  const int b  = row0 / rows_per_batch;   // 16-row tile never crosses batch
  const int n0 = row0 % rows_per_batch;
#pragma unroll
  for (int v = 0; v < 8; ++v) {
    int rloc = v + 8 * half;
    float ns = ssq[rloc][hb] + ssq[rloc][hb + 1];
    float scale = 1.0f / fmaxf(sqrtf(ns), 1e-12f);
    size_t base = (((size_t)b * H_ + h) * rows_per_batch + (n0 + rloc)) * D_;
    out_lat[base + d0]      = f2bf(v0[v] * scale);
    out_lat[base + d0 + 16] = f2bf(v1[v] * scale);
  }
}

// ------------------------- interaction (cosine sims) + masked streaming LSE
// Block = 4 waves, each wave a 16-row i tile. The whole (b,h) aa-latent panel
// (512 x 64 bf16 = 64KB) is DMA'd once into LDS by the Tensor Data Mover;
// the j-loop then feeds WMMA B-fragments from LDS (ds_load_b128).
__global__ void k_inter_lse(const unsigned short* __restrict__ seq_lat,
                            const unsigned short* __restrict__ aa_lat,
                            const unsigned char* __restrict__ aa_mask,
                            float* __restrict__ lse_out) {
  __shared__ __align__(16) unsigned short latB[M_ * D_];   // 64KB B panel

  const int bh   = blockIdx.x;
  const int b    = bh >> 5;
  const int h    = bh & 31;
  const int wave = threadIdx.x >> 5;
  const int lane = threadIdx.x & 31;
  const int half = lane >> 4;
  const int lc   = lane & 15;
  const int i0   = blockIdx.y * 64 + wave * 16;

  // --- TDM: stage aa_lat(b,h) panel into LDS (one DMA, wave 0 issues) ---
  if (threadIdx.x < 32) {
    unsigned long long gaddr =
        (unsigned long long)(const void*)(aa_lat + (size_t)bh * M_ * D_);
    unsigned lds_base = (unsigned)(unsigned long long)(void*)latB;
    const unsigned NELEM = (unsigned)(M_ * D_);            // 32768 bf16 elements

    v4u g0;
    g0[0] = 1u;                                            // count=1 (valid D#)
    g0[1] = lds_base;                                      // lds_addr (bytes)
    g0[2] = (unsigned)(gaddr & 0xFFFFFFFFull);             // global_addr[31:0]
    g0[3] = (unsigned)((gaddr >> 32) & 0x01FFFFFFull) | (2u << 30); // addr hi + type=2

    v8i g1;
    g1[0] = 0x00010000;                    // wg_mask=0, data_size=1 (2 bytes)
    g1[1] = (int)((NELEM & 0xFFFFu) << 16);        // tensor_dim0[15:0]
    g1[2] = (int)(((NELEM >> 16) & 0xFFFFu) | (1u << 16)); // dim0 hi | tensor_dim1=1
    g1[3] = (int)((NELEM & 0xFFFFu) << 16);        // tile_dim0 = 32768 (1-D tile)
    g1[4] = 0;                                     // tile_dim1/2 unused
    g1[5] = (int)NELEM;                            // tensor_dim0_stride lo
    g1[6] = 0;
    g1[7] = 0;

    v4i g2 = {};
    v4i g3 = {};
#if __clang_major__ >= 23
    v8i g4 = {};
    __builtin_amdgcn_tensor_load_to_lds(g0, g1, g2, g3, g4, 0);
#else
    __builtin_amdgcn_tensor_load_to_lds(g0, g1, g2, g3, 0);
#endif
    __builtin_amdgcn_s_wait_tensorcnt(0);
  }
  __syncthreads();

  // A fragments (seq rows i0..i0+15, K = d 0..63) preloaded once
  const __bf16* ap = (const __bf16*)seq_lat + ((size_t)bh * N_ + i0 + lc) * D_ + half * 8;
  v16bf a0 = cat16(*(const v8bf*)(ap),      *(const v8bf*)(ap + 16));
  v16bf a1 = cat16(*(const v8bf*)(ap + 32), *(const v8bf*)(ap + 48));

  const unsigned char* mrow = aa_mask + (size_t)b * M_;

  float mrun[8], srun[8];
#pragma unroll
  for (int v = 0; v < 8; ++v) { mrun[v] = -3.402823466e38f; srun[v] = 0.0f; }

  for (int j0 = 0; j0 < M_; j0 += 16) {
    const __bf16* bp = (const __bf16*)&latB[(j0 + lc) * D_ + half * 8];
    v16bf b0 = cat16(*(const v8bf*)(bp),      *(const v8bf*)(bp + 16));
    v16bf b1 = cat16(*(const v8bf*)(bp + 32), *(const v8bf*)(bp + 48));

    v8f acc = {};
    acc = __builtin_amdgcn_wmma_f32_16x16x32_bf16(false, a0, false, b0,
                                                  (short)0, acc, false, false);
    acc = __builtin_amdgcn_wmma_f32_16x16x32_bf16(false, a1, false, b1,
                                                  (short)0, acc, false, false);

    const bool mk = mrow[j0 + lc] != 0;          // this lane's column j
    float sc[8], tmax[8];
#pragma unroll
    for (int v = 0; v < 8; ++v) { sc[v] = mk ? -3.402823466e38f : acc[v]; tmax[v] = sc[v]; }

    // tile row-max over the 16 columns (lanes of this half)
#pragma unroll
    for (int m = 1; m <= 8; m <<= 1)
#pragma unroll
      for (int v = 0; v < 8; ++v) tmax[v] = fmaxf(tmax[v], __shfl_xor(tmax[v], m, 32));

    float mnew[8], p[8];
#pragma unroll
    for (int v = 0; v < 8; ++v) {
      mnew[v] = fmaxf(mrun[v], tmax[v]);
      p[v]    = __expf(sc[v] - mnew[v]);
    }
#pragma unroll
    for (int m = 1; m <= 8; m <<= 1)
#pragma unroll
      for (int v = 0; v < 8; ++v) p[v] += __shfl_xor(p[v], m, 32);
#pragma unroll
    for (int v = 0; v < 8; ++v) {
      srun[v] = srun[v] * __expf(mrun[v] - mnew[v]) + p[v];
      mrun[v] = mnew[v];
    }
  }

  if (lc == 0) {                                  // lanes 0 and 16 write 8 rows each
#pragma unroll
    for (int v = 0; v < 8; ++v)
      lse_out[((size_t)b * N_ + i0 + v + 8 * half) * H_ + h] =
          mrun[v] + __logf(srun[v]);
  }
}

// ------------------------------------------------ contextual gating (tiny)
__global__ void k_gating(const float* __restrict__ ctx, const float* __restrict__ ctx_w,
                         const float* __restrict__ ctx_b, const float* __restrict__ tlw,
                         float* __restrict__ w_eff) {
  const int b = blockIdx.x;
  for (int o = threadIdx.x; o < H_ * H_; o += blockDim.x) {
    float s = ctx_b[o];
    const float* c = ctx + (size_t)b * CTX_;
    for (int k = 0; k < CTX_; ++k) s += c[k] * ctx_w[(size_t)k * (H_ * H_) + o];
    float g = 1.0f / (1.0f + __expf(-s));
    w_eff[(size_t)b * (H_ * H_) + o] = tlw[o] * g;   // o = d*32 + e, row-major (H,H)
  }
}

// ---------------------------------------------- logits @ pred_w + softplus
__global__ void k_final(const float* __restrict__ lse, const float* __restrict__ w_eff,
                        const float* __restrict__ pred_w, const float* __restrict__ pred_b,
                        float* __restrict__ out) {
  const int idx = blockIdx.x * blockDim.x + threadIdx.x;   // over B*N
  if (idx >= B_ * N_) return;
  const int b = idx / N_;
  const float* r = lse + (size_t)idx * H_;
  const float* W = w_eff + (size_t)b * (H_ * H_);
  float x = pred_b[0];
#pragma unroll 4
  for (int e = 0; e < H_; ++e) {
    float a = 0.0f;
    for (int d = 0; d < H_; ++d) a += r[d] * W[d * H_ + e];
    x += a * pred_w[e];
  }
  out[idx] = fmaxf(x, 0.0f) + log1pf(__expf(-fabsf(x)));   // stable softplus
}

// ---------------------------------------------------------------- dispatch
extern "C" void kernel_launch(void* const* d_in, const int* in_sizes, int n_in,
                              void* d_out, int out_size, void* d_ws, size_t ws_size,
                              hipStream_t stream) {
  (void)in_sizes; (void)n_in; (void)out_size; (void)ws_size;
  const float* seq_embed = (const float*)d_in[0];
  const float* aa_embed  = (const float*)d_in[1];
  const float* ctx       = (const float*)d_in[2];
  const float* seq_w     = (const float*)d_in[3];
  const float* seq_b     = (const float*)d_in[4];
  const float* aa_w      = (const float*)d_in[5];
  const float* aa_b      = (const float*)d_in[6];
  const float* tlw       = (const float*)d_in[7];
  const float* ctx_w     = (const float*)d_in[8];
  const float* ctx_b     = (const float*)d_in[9];
  const float* pred_w    = (const float*)d_in[10];
  const float* pred_b    = (const float*)d_in[11];
  const unsigned char* aa_mask = (const unsigned char*)d_in[12];
  float* out = (float*)d_out;

  char* ws = (char*)d_ws;
  size_t off = 0;
  auto carve = [&](size_t bytes) -> char* {
    char* p = ws + off;
    off += (bytes + 255) & ~(size_t)255;
    return p;
  };
  unsigned short* seqE_bf = (unsigned short*)carve((size_t)B_ * N_ * ENF2 * 2);
  unsigned short* aaE_bf  = (unsigned short*)carve((size_t)B_ * M_ * AAD * 2);
  unsigned short* seqWT   = (unsigned short*)carve((size_t)ENF2 * INNER * 2);
  unsigned short* aaWT    = (unsigned short*)carve((size_t)AAD * INNER * 2);
  unsigned short* seq_lat = (unsigned short*)carve((size_t)B_ * H_ * N_ * D_ * 2);
  unsigned short* aa_lat  = (unsigned short*)carve((size_t)B_ * H_ * M_ * D_ * 2);
  float* lse   = (float*)carve((size_t)B_ * N_ * H_ * 4);
  float* w_eff = (float*)carve((size_t)B_ * H_ * H_ * 4);

  // 1) bf16 conversions (weights transposed so B-fragments are K-contiguous)
  k_f32_to_bf16 <<<dim3(4096), 256, 0, stream>>>(seq_embed, seqE_bf, (size_t)B_ * N_ * ENF2);
  k_f32_to_bf16 <<<dim3(2048), 256, 0, stream>>>(aa_embed,  aaE_bf,  (size_t)B_ * M_ * AAD);
  k_f32_to_bf16_T<<<dim3(4096), 256, 0, stream>>>(seq_w, seqWT, ENF2, INNER);
  k_f32_to_bf16_T<<<dim3(2048), 256, 0, stream>>>(aa_w,  aaWT,  AAD,  INNER);

  // 2) projections + bias + per-head l2norm -> bf16 latents (b,h,row,d)
  k_gemm_bias_l2norm<<<dim3((B_ * N_) / 16, INNER / 256), 256, 0, stream>>>(
      (const __bf16*)seqE_bf, (const __bf16*)seqWT, seq_b, seq_lat, N_, ENF2);
  k_gemm_bias_l2norm<<<dim3((B_ * M_) / 16, INNER / 256), 256, 0, stream>>>(
      (const __bf16*)aaE_bf, (const __bf16*)aaWT, aa_b, aa_lat, M_, AAD);

  // 3) interaction scores + masked streaming logsumexp -> (b,n,h)
  k_inter_lse<<<dim3(B_ * H_, N_ / 64), 128, 0, stream>>>(
      seq_lat, aa_lat, aa_mask, lse);

  // 4) gating + logits + softplus
  k_gating<<<dim3(B_), 256, 0, stream>>>(ctx, ctx_w, ctx_b, tlw, w_eff);
  k_final <<<dim3((B_ * N_ + 255) / 256), 256, 0, stream>>>(lse, w_eff, pred_w, pred_b, out);
}
